// ConvGRUForecaster_81183471829634
// MI455X (gfx1250) — matrix operand order repaired and checked by reading, not compiled, not run on hardware
//
#include <hip/hip_runtime.h>

// ---------------------------------------------------------------------------
// ConvGRUForecaster for MI455X (gfx1250, wave32, WMMA)
// N=20000 nodes, E=320000 edges, IN=128, HID=256, OUT=32
// ---------------------------------------------------------------------------

#define NN   20000
#define EE   320000
#define INF_ 128
#define HID_ 256
#define H3_  768
#define OUT_ 32

typedef __attribute__((ext_vector_type(16))) __bf16 bf16x16;
typedef __attribute__((ext_vector_type(8)))  float  f32x8;

struct __attribute__((aligned(16))) U128 { unsigned int x, y, z, w; };

union ABFrag {
    bf16x16 v;
    U128    q[2];
};

__device__ __forceinline__ unsigned short f2bf(float f) {
    unsigned int u = __float_as_uint(f);
    unsigned int r = u + 0x7FFFu + ((u >> 16) & 1u);   // round-to-nearest-even
    return (unsigned short)(r >> 16);
}

__device__ __forceinline__ float sigmoidf_(float x) {
    return 1.0f / (1.0f + __expf(-x));
}

// ---------------------------------------------------------------------------
// Degree / normalization kernels
// ---------------------------------------------------------------------------
__global__ void deg_init_k(float* deg, int n) {
    int i = blockIdx.x * blockDim.x + threadIdx.x;
    if (i < n) deg[i] = 1.0f;   // self-loop contributes 1
}

__global__ void deg_count_k(const int* dst, float* deg, int e) {
    int i = blockIdx.x * blockDim.x + threadIdx.x;
    if (i < e) atomicAdd(&deg[dst[i]], 1.0f);
}

__global__ void dinv_k(const float* deg, float* dinv, int n) {
    int i = blockIdx.x * blockDim.x + threadIdx.x;
    if (i < n) dinv[i] = __frsqrt_rn(deg[i]);   // deg >= 1 always
}

// ---------------------------------------------------------------------------
// f32 -> bf16 convert (optional fused ReLU)
// ---------------------------------------------------------------------------
__global__ void f2bf_k(const float* __restrict__ in, unsigned short* __restrict__ out,
                       long long n, int relu) {
    long long i = (long long)blockIdx.x * blockDim.x + threadIdx.x;
    if (i < n) {
        float v = in[i];
        if (relu) v = fmaxf(v, 0.0f);
        out[i] = f2bf(v);
    }
}

// ---------------------------------------------------------------------------
// Pack B (f32 [K,N] row-major, or transposed source [N,K]) into the WMMA
// B-fragment memory layout: per (ntile, kblock) region of 1KB, lane L owns
// 32 contiguous bytes; dword j of lane L packs B[k][n], B[k+1][n] with
//   k = kb*32 + 8*(L>>4) + 2j + (j<4 ? 0 : 8),  n = tn*16 + (L&15)
// so the GEMM inner loop does two b128 loads per lane per K-step per tile.
// ---------------------------------------------------------------------------
__global__ void pack_b_k(const float* __restrict__ src, unsigned short* __restrict__ dst,
                         int K, int N, int trans) {
    int kblocks = K >> 5;
    int total = kblocks * (N >> 4) * 256;          // dwords
    int tid = blockIdx.x * blockDim.x + threadIdx.x;
    if (tid >= total) return;
    int j  = tid & 7;
    int L  = (tid >> 3) & 31;
    int r  = tid >> 8;
    int kb = r % kblocks;
    int tn = r / kblocks;
    int k  = kb * 32 + ((L < 16) ? 0 : 8) + 2 * j + ((j < 4) ? 0 : 8);
    int n  = tn * 16 + (L & 15);
    float lo, hi;
    if (trans) {  // src is [N,K] row-major; logical B[k][n] = src[n*K + k]
        lo = src[(long long)n * K + k];
        hi = src[(long long)n * K + k + 1];
    } else {      // src is [K,N] row-major
        lo = src[(long long)k * N + n];
        hi = src[(long long)(k + 1) * N + n];
    }
    ((unsigned int*)dst)[tid] = (unsigned int)f2bf(lo) | ((unsigned int)f2bf(hi) << 16);
}

// ---------------------------------------------------------------------------
// WMMA BF16 GEMM with register blocking:
//   C[M,Ncols] = A[M,K]_bf16 @ Bpacked  (+bias) (f32 out)
// Each wave computes an (MT*16) x (NT*16) macro-tile: per K-step of 32,
// MT A-fragment loads + NT B-fragment loads feed MT*NT
// v_wmma_f32_16x16x32_bf16 ops (1.5 b128 loads per WMMA at 2x4).
// M%(16*MT)==0, Ncols%(16*NT)==0, K%32==0 guaranteed by launcher.
// ---------------------------------------------------------------------------
template <int MT, int NT>
__global__ __launch_bounds__(256) void wmma_gemm_k(
        const unsigned short* __restrict__ A,
        const unsigned short* __restrict__ Bp,
        const float* __restrict__ bias,
        float* __restrict__ C,
        int M, int Ncols, int K, int relu) {
    int ntiles  = Ncols >> 4;
    int mtiles  = M >> 4;
    int ngroups = ntiles / NT;
    int groups  = (mtiles / MT) * ngroups;
    int w       = blockIdx.x * (blockDim.x >> 5) + (threadIdx.x >> 5);
    if (w >= groups) return;               // wave-uniform exit (EXEC stays full)
    int lane    = threadIdx.x & 31;
    int gM      = w / ngroups;
    int gN      = w - gM * ngroups;
    int kblocks = K >> 5;
    int hiHalf  = lane >> 4;               // 0: lanes 0-15, 1: lanes 16-31

    f32x8 acc[MT][NT];
    #pragma unroll
    for (int mt = 0; mt < MT; ++mt)
        #pragma unroll
        for (int nt = 0; nt < NT; ++nt)
            acc[mt][nt] = (f32x8){0.f, 0.f, 0.f, 0.f, 0.f, 0.f, 0.f, 0.f};

    const unsigned short* aBase[MT];
    #pragma unroll
    for (int mt = 0; mt < MT; ++mt)
        aBase[mt] = A + (long long)((gM * MT + mt) * 16 + (lane & 15)) * K + (hiHalf << 3);

    const U128* bBase[NT];
    #pragma unroll
    for (int nt = 0; nt < NT; ++nt)
        bBase[nt] = (const U128*)(Bp + (long long)((gN * NT + nt) * kblocks) * 512) + lane * 2;

    for (int kb = 0; kb < kblocks; ++kb) {
        ABFrag a[MT], b[NT];
        #pragma unroll
        for (int mt = 0; mt < MT; ++mt) {
            const unsigned short* ap = aBase[mt] + kb * 32;
            a[mt].q[0] = *(const U128*)(ap);         // K = base .. base+7
            a[mt].q[1] = *(const U128*)(ap + 16);    // K = base+16 .. base+23
        }
        #pragma unroll
        for (int nt = 0; nt < NT; ++nt) {
            const U128* bp = bBase[nt] + (long long)kb * 64;   // 64 U128 per 1KB region
            b[nt].q[0] = bp[0];
            b[nt].q[1] = bp[1];
        }
        #pragma unroll
        for (int mt = 0; mt < MT; ++mt)
            #pragma unroll
            for (int nt = 0; nt < NT; ++nt)
                acc[mt][nt] = __builtin_amdgcn_wmma_f32_16x16x32_bf16(
                                  false, a[mt].v, false, b[nt].v,
                                  (short)0, acc[mt][nt], false, false);
    }

    #pragma unroll
    for (int mt = 0; mt < MT; ++mt) {
        int rowBase = (gM * MT + mt) * 16 + (hiHalf << 3);
        #pragma unroll
        for (int nt = 0; nt < NT; ++nt) {
            int col = (gN * NT + nt) * 16 + (lane & 15);
            float bv = (bias != nullptr) ? bias[col] : 0.0f;
            #pragma unroll
            for (int r = 0; r < 8; ++r) {
                float v = acc[mt][nt][r] + bv;
                if (relu) v = fmaxf(v, 0.0f);
                C[(long long)(rowBase + r) * Ncols + col] = v;
            }
        }
    }
}

// ---------------------------------------------------------------------------
// GCN aggregation: out[i][c] = dinv[i]^2 * xw[i][c] + bias[c]  (self loop)
// then edge scatter: out[d][c] += dinv[s]*dinv[d] * xw[s][c]
// ---------------------------------------------------------------------------
__global__ void agg_init_k(const float* __restrict__ dinv, const float* __restrict__ xw,
                           const float* __restrict__ bias, float* __restrict__ out,
                           int n, int h) {
    long long i = (long long)blockIdx.x * blockDim.x + threadIdx.x;
    if (i >= (long long)n * h) return;
    int node = (int)(i / h);
    int c    = (int)(i - (long long)node * h);
    float dv = dinv[node];
    out[i] = dv * dv * xw[i] + bias[c];
}

__global__ void edge_scatter_k(const int* __restrict__ src, const int* __restrict__ dst,
                               const float* __restrict__ dinv, const float* __restrict__ xw,
                               float* __restrict__ out, int e, int h) {
    int edge = blockIdx.x * (blockDim.x >> 5) + (threadIdx.x >> 5);
    if (edge >= e) return;
    int lane = threadIdx.x & 31;
    int s = src[edge];
    int d = dst[edge];
    float nrm = dinv[s] * dinv[d];
    const float4* xs = (const float4*)(xw + (long long)s * h);
    float* od = out + (long long)d * h;
    for (int j = lane; j < (h >> 2); j += 32) {
        float4 v = xs[j];
        atomicAdd(od + 4 * j + 0, nrm * v.x);
        atomicAdd(od + 4 * j + 1, nrm * v.y);
        atomicAdd(od + 4 * j + 2, nrm * v.z);
        atomicAdd(od + 4 * j + 3, nrm * v.w);
    }
}

// ---------------------------------------------------------------------------
// Sequential GRU scan (torch semantics). One persistent 768-thread workgroup
// (24 wave32s). h[256] and gate pre-activations hp[768] live in LDS; each
// thread owns one row of Whh (streamed from L2 every step).
//   hp = Whh @ h + bhh ; r=sig(xr+hr); z=sig(xz+hz); n=tanh(xn+r*hn)
//   h' = (1-z)*n + z*h
// ---------------------------------------------------------------------------
__global__ __launch_bounds__(768, 1) void gru_scan_k(
        const float* __restrict__ xp,      // [T, 768]
        const float* __restrict__ Whh,     // [768, 256]
        const float* __restrict__ bhh,     // [768]
        float* __restrict__ out,           // [T, 256]
        int T) {
    __shared__ float h[HID_];
    __shared__ float hp[H3_];
    int t = threadIdx.x;
    if (t < HID_) h[t] = 0.0f;
    __syncthreads();

    const float4* wrow = (const float4*)(Whh + (long long)t * HID_);
    float bh = bhh[t];

    for (int step = 0; step < T; ++step) {
        float acc = bh;
        const float4* hv = (const float4*)h;
        #pragma unroll 8
        for (int j = 0; j < HID_ / 4; ++j) {
            float4 wv = wrow[j];
            float4 xv = hv[j];
            acc += wv.x * xv.x + wv.y * xv.y + wv.z * xv.z + wv.w * xv.w;
        }
        hp[t] = acc;
        __syncthreads();
        if (t < HID_) {
            const float* xr = xp + (long long)step * H3_;
            float r  = sigmoidf_(xr[t]            + hp[t]);
            float z  = sigmoidf_(xr[HID_ + t]     + hp[HID_ + t]);
            float nc = tanhf(xr[2 * HID_ + t] + r * hp[2 * HID_ + t]);
            float hn = (1.0f - z) * nc + z * h[t];
            h[t] = hn;
            out[(long long)step * HID_ + t] = hn;
        }
        __syncthreads();
    }
}

// ---------------------------------------------------------------------------
// Host-side orchestration
// ---------------------------------------------------------------------------
static inline size_t align256(size_t x) { return (x + 255) & ~(size_t)255; }
static inline int cdiv_i(long long a, long long b) { return (int)((a + b - 1) / b); }

extern "C" void kernel_launch(void* const* d_in, const int* in_sizes, int n_in,
                              void* d_out, int out_size, void* d_ws, size_t ws_size,
                              hipStream_t stream) {
    const float* x    = (const float*)d_in[0];
    const int*   ei   = (const int*)d_in[1];
    const float* W1   = (const float*)d_in[2];
    const float* b1   = (const float*)d_in[3];
    const float* W2   = (const float*)d_in[4];
    const float* b2   = (const float*)d_in[5];
    const float* Wih1 = (const float*)d_in[6];
    const float* Whh1 = (const float*)d_in[7];
    const float* bih1 = (const float*)d_in[8];
    const float* bhh1 = (const float*)d_in[9];
    const float* Wih2 = (const float*)d_in[10];
    const float* Whh2 = (const float*)d_in[11];
    const float* bih2 = (const float*)d_in[12];
    const float* bhh2 = (const float*)d_in[13];
    const float* Wl   = (const float*)d_in[14];
    const float* bl   = (const float*)d_in[15];

    const int N = NN, E = EE;
    const int* srcIdx = ei;
    const int* dstIdx = ei + E;

    // ---- carve workspace ----
    char* p = (char*)d_ws;
    auto carve = [&](size_t bytes) { char* r = p; p += align256(bytes); return r; };
    float*          deg   = (float*)         carve((size_t)N * 4);
    float*          dinv  = (float*)         carve((size_t)N * 4);
    unsigned short* x_bf  = (unsigned short*)carve((size_t)N * INF_ * 2);
    unsigned short* W1p   = (unsigned short*)carve((size_t)INF_ * HID_ * 2);
    unsigned short* W2p   = (unsigned short*)carve((size_t)HID_ * HID_ * 2);
    unsigned short* Wt1p  = (unsigned short*)carve((size_t)HID_ * H3_ * 2);
    unsigned short* Wt2p  = (unsigned short*)carve((size_t)HID_ * H3_ * 2);
    unsigned short* Wlp   = (unsigned short*)carve((size_t)HID_ * OUT_ * 2);
    float*          xw    = (float*)         carve((size_t)N * HID_ * 4);
    float*          hpre  = (float*)         carve((size_t)N * HID_ * 4);
    unsigned short* hbf   = (unsigned short*)carve((size_t)N * HID_ * 2);
    float*          xp    = (float*)         carve((size_t)N * H3_ * 4);
    float*          gout  = (float*)         carve((size_t)N * HID_ * 4);

    const int TB = 256;

    // ---- degree / normalization ----
    deg_init_k<<<cdiv_i(N, TB), TB, 0, stream>>>(deg, N);
    deg_count_k<<<cdiv_i(E, TB), TB, 0, stream>>>(dstIdx, deg, E);
    dinv_k<<<cdiv_i(N, TB), TB, 0, stream>>>(deg, dinv, N);

    // ---- precision conversion + WMMA weight packing ----
    f2bf_k<<<cdiv_i((long long)N * INF_, TB), TB, 0, stream>>>(x, x_bf, (long long)N * INF_, 0);
    pack_b_k<<<cdiv_i((INF_ / 32) * (HID_ / 16) * 256, TB), TB, 0, stream>>>(W1, W1p, INF_, HID_, 0);
    pack_b_k<<<cdiv_i((HID_ / 32) * (HID_ / 16) * 256, TB), TB, 0, stream>>>(W2, W2p, HID_, HID_, 0);
    pack_b_k<<<cdiv_i((HID_ / 32) * (H3_ / 16) * 256, TB), TB, 0, stream>>>(Wih1, Wt1p, HID_, H3_, 1);
    pack_b_k<<<cdiv_i((HID_ / 32) * (H3_ / 16) * 256, TB), TB, 0, stream>>>(Wih2, Wt2p, HID_, H3_, 1);
    pack_b_k<<<cdiv_i((HID_ / 32) * (OUT_ / 16) * 256, TB), TB, 0, stream>>>(Wl, Wlp, HID_, OUT_, 0);

    // 2x4 register-blocked GEMM for wide outputs (Ncols % 64 == 0, M % 32 == 0)
    auto gemm24 = [&](const unsigned short* A, const unsigned short* Bp, const float* bias,
                      float* C, int M, int Ncols, int K, int relu) {
        int groups = (M / 32) * (Ncols / 64);
        wmma_gemm_k<2, 4><<<cdiv_i(groups, 8), 256, 0, stream>>>(A, Bp, bias, C, M, Ncols, K, relu);
    };
    // 2x2 for the narrow final linear (Ncols % 32 == 0)
    auto gemm22 = [&](const unsigned short* A, const unsigned short* Bp, const float* bias,
                      float* C, int M, int Ncols, int K, int relu) {
        int groups = (M / 32) * (Ncols / 32);
        wmma_gemm_k<2, 2><<<cdiv_i(groups, 8), 256, 0, stream>>>(A, Bp, bias, C, M, Ncols, K, relu);
    };

    // ---- GCN layer 1 ----
    gemm24(x_bf, W1p, nullptr, xw, N, HID_, INF_, 0);
    agg_init_k<<<cdiv_i((long long)N * HID_, TB), TB, 0, stream>>>(dinv, xw, b1, hpre, N, HID_);
    edge_scatter_k<<<cdiv_i(E, 8), 256, 0, stream>>>(srcIdx, dstIdx, dinv, xw, hpre, E, HID_);
    f2bf_k<<<cdiv_i((long long)N * HID_, TB), TB, 0, stream>>>(hpre, hbf, (long long)N * HID_, 1);

    // ---- GCN layer 2 ----
    gemm24(hbf, W2p, nullptr, xw, N, HID_, HID_, 0);
    agg_init_k<<<cdiv_i((long long)N * HID_, TB), TB, 0, stream>>>(dinv, xw, b2, hpre, N, HID_);
    edge_scatter_k<<<cdiv_i(E, 8), 256, 0, stream>>>(srcIdx, dstIdx, dinv, xw, hpre, E, HID_);
    f2bf_k<<<cdiv_i((long long)N * HID_, TB), TB, 0, stream>>>(hpre, hbf, (long long)N * HID_, 1);

    // ---- GRU layer 1 ----
    gemm24(hbf, Wt1p, bih1, xp, N, H3_, HID_, 0);
    gru_scan_k<<<1, H3_, 0, stream>>>(xp, Whh1, bhh1, gout, N);
    f2bf_k<<<cdiv_i((long long)N * HID_, TB), TB, 0, stream>>>(gout, hbf, (long long)N * HID_, 0);

    // ---- GRU layer 2 ----
    gemm24(hbf, Wt2p, bih2, xp, N, H3_, HID_, 0);
    gru_scan_k<<<1, H3_, 0, stream>>>(xp, Whh2, bhh2, gout, N);
    f2bf_k<<<cdiv_i((long long)N * HID_, TB), TB, 0, stream>>>(gout, hbf, (long long)N * HID_, 0);

    // ---- final linear -> d_out ----
    gemm22(hbf, Wlp, bl, (float*)d_out, N, OUT_, HID_, 0);
}
